// MultiHeadSelfAttention_49649821942085
// MI455X (gfx1250) — compile-verified, matrix operand order checked by reading, not run on hardware
//
#include <hip/hip_runtime.h>
#include <stdint.h>

// ---------------------------------------------------------------------------
// Multi-head self attention for MI455X (gfx1250), bf16 WMMA pipeline with
// CDNA5 async-to-LDS staging.
//   B=2, S=2048, D=1024, H=16, DEPTH=64
// ---------------------------------------------------------------------------

typedef __attribute__((ext_vector_type(16))) __bf16 v16bf;
typedef __attribute__((ext_vector_type(8)))  float  v8f;

#define B_     2
#define S_     2048
#define D_     1024
#define H_     16
#define DEPTH_ 64
#define M_     (B_ * S_)   // 4096 rows of the "token" matrix

// round-to-nearest-even float -> bf16 (stored as ushort)
__device__ __forceinline__ unsigned short f2bf(float x) {
    union { float f; uint32_t u; } v; v.f = x;
    uint32_t r = v.u + 0x7FFFu + ((v.u >> 16) & 1u);
    return (unsigned short)(r >> 16);
}

__device__ __forceinline__ v8f vzero8() {
    v8f z;
#pragma unroll
    for (int i = 0; i < 8; ++i) z[i] = 0.0f;
    return z;
}

// CDNA5 async global->LDS copy (16B per lane), tracked on ASYNCcnt.
__device__ __forceinline__ void async_ld_b128(unsigned lds_addr, const void* gptr) {
    asm volatile("global_load_async_to_lds_b128 %0, %1, off"
                 :: "v"(lds_addr), "v"((unsigned long long)(uintptr_t)gptr)
                 : "memory");
}
__device__ __forceinline__ void wait_async0() {
    asm volatile("s_wait_asynccnt 0x0" ::: "memory");
}

// A/B operand fragment: 16 bf16 per lane = two 16-byte chunks
union FragAB { v16bf v; uint4 q[2]; };

// ---------------------------------------------------------------------------
// fp32 -> bf16 elementwise convert
// ---------------------------------------------------------------------------
__global__ void cvt_bf16(const float* __restrict__ in,
                         unsigned short* __restrict__ out, int n) {
    int i = blockIdx.x * blockDim.x + threadIdx.x;
    if (i < n) out[i] = f2bf(in[i]);
}

// ---------------------------------------------------------------------------
// fp32 W[K=1024][N=1024] -> bf16 Wt[N][K]  (tiled transpose through LDS)
// launch: block (32,8), grid (32,32)
// ---------------------------------------------------------------------------
__global__ void cvt_wT(const float* __restrict__ w,
                       unsigned short* __restrict__ wt) {
    __shared__ float tile[32][33];
    const int k0 = blockIdx.x * 32;
    const int n0 = blockIdx.y * 32;
    const int tx = threadIdx.x;
    const int ty = threadIdx.y;
#pragma unroll
    for (int i = 0; i < 32; i += 8)
        tile[ty + i][tx] = w[(size_t)(k0 + ty + i) * D_ + n0 + tx];
    __syncthreads();
#pragma unroll
    for (int i = 0; i < 32; i += 8)
        wt[(size_t)(n0 + ty + i) * D_ + k0 + tx] = f2bf(tile[tx][ty + i]);
}

// ---------------------------------------------------------------------------
// bf16 GEMM:  C[M][N] = A[M][K] * BT[N][K]^T.
// Block tile 128x128, 256 threads = 8 waves; wave tile 32(M) x 64(N).
// K-slab 64, double-buffered LDS, async-to-LDS staging (1 barrier / slab).
// ---------------------------------------------------------------------------
#define GT_M 128
#define GT_N 128
#define GT_K 64
#define LDSW 72   // padded LDS stride (elements); 144 bytes, 16B aligned

__global__ __launch_bounds__(256)
void gemm_bf16_wmma(const unsigned short* __restrict__ A,
                    const unsigned short* __restrict__ BT,
                    float* __restrict__ Cf,
                    unsigned short* __restrict__ Cb,
                    int M, int N, int K, int store_f32) {
    __shared__ unsigned short sA[2][GT_M * LDSW];
    __shared__ unsigned short sB[2][GT_N * LDSW];

    const int tid  = threadIdx.x;
    const int wave = tid >> 5;
    const int lane = tid & 31;
    const int lm   = lane & 15;   // row/col within 16
    const int g    = lane >> 4;   // lane-half selects K sub-range
    const int wm   = wave >> 1;   // 0..3 -> M offset 32*wm
    const int wn   = wave & 1;    // 0..1 -> N offset 64*wn
    const int m0   = blockIdx.y * GT_M;
    const int n0   = blockIdx.x * GT_N;

    v8f acc[2][4];
#pragma unroll
    for (int i = 0; i < 2; ++i)
#pragma unroll
        for (int j = 0; j < 4; ++j) acc[i][j] = vzero8();

    const int lr = tid >> 1;          // 0..127 tile row
    const int lc = (tid & 1) * 32;    // 0 / 32 column chunk (32 elems = 4x16B)

    // stage one 128x64 A tile + 128x64 B tile into LDS buffer `buf`
    auto stage = [&](int buf, int k0) {
        const unsigned short* ga = A  + (size_t)(m0 + lr) * K + k0 + lc;
        const unsigned short* gb = BT + (size_t)(n0 + lr) * K + k0 + lc;
        unsigned la = (unsigned)(uintptr_t)&sA[buf][lr * LDSW + lc];
        unsigned lb = (unsigned)(uintptr_t)&sB[buf][lr * LDSW + lc];
#pragma unroll
        for (int c = 0; c < 4; ++c) {
            async_ld_b128(la + c * 16, ga + c * 8);
            async_ld_b128(lb + c * 16, gb + c * 8);
        }
    };

    stage(0, 0);
    wait_async0();
    __syncthreads();

    const int nIter = K / GT_K;
    for (int it = 0; it < nIter; ++it) {
        const int buf = it & 1;
        if (it + 1 < nIter) stage(buf ^ 1, (it + 1) * GT_K);

#pragma unroll
        for (int ks = 0; ks < GT_K; ks += 32) {
            FragAB af[2];
#pragma unroll
            for (int am = 0; am < 2; ++am) {
                const int base = (wm * 32 + am * 16 + lm) * LDSW + ks + g * 8;
                af[am].q[0] = *(const uint4*)&sA[buf][base];
                af[am].q[1] = *(const uint4*)&sA[buf][base + 16];
            }
#pragma unroll
            for (int bn = 0; bn < 4; ++bn) {
                FragAB bf;
                const int base = (wn * 64 + bn * 16 + lm) * LDSW + ks + g * 8;
                bf.q[0] = *(const uint4*)&sB[buf][base];
                bf.q[1] = *(const uint4*)&sB[buf][base + 16];
#pragma unroll
                for (int am = 0; am < 2; ++am) {
                    acc[am][bn] = __builtin_amdgcn_wmma_f32_16x16x32_bf16(
                        false, af[am].v, false, bf.v,
                        (short)0, acc[am][bn], false, false);
                }
            }
        }

        wait_async0();      // own async loads into buf^1 done
        __syncthreads();    // everyone's loads done; reads of buf finished
    }

    // epilogue: C/D layout -> row m = vr + 8*g, col n = lm
#pragma unroll
    for (int am = 0; am < 2; ++am)
#pragma unroll
        for (int bn = 0; bn < 4; ++bn)
#pragma unroll
            for (int vr = 0; vr < 8; ++vr) {
                const int row = m0 + wm * 32 + am * 16 + vr + 8 * g;
                const int col = n0 + wn * 64 + bn * 16 + lm;
                if (store_f32) Cf[(size_t)row * N + col] = acc[am][bn][vr];
                else           Cb[(size_t)row * N + col] = f2bf(acc[am][bn][vr]);
            }
}

// ---------------------------------------------------------------------------
// Flash attention (online softmax) with bf16 WMMA.
// grid: (S/128, H, B); block: 256 threads = 8 waves; wave owns 16 q-rows.
// Key chunks of 64; K staged row-major via async-to-LDS, V staged transposed.
// ---------------------------------------------------------------------------
#define KC    64
#define SKW   72   // padded LDS stride (elements), 144 bytes

__global__ __launch_bounds__(256)
void attn_wmma(const unsigned short* __restrict__ Q,
               const unsigned short* __restrict__ Kmat,
               const unsigned short* __restrict__ V,
               unsigned short* __restrict__ Ctx) {
    __shared__ unsigned short sK [KC * SKW];        // [key][depth]
    __shared__ unsigned short sVT[DEPTH_ * SKW];    // [depth][key]
    __shared__ unsigned short sP [8][16 * SKW];     // per-wave probs [qrow][key]

    const int tid  = threadIdx.x;
    const int wave = tid >> 5;
    const int lane = tid & 31;
    const int lm   = lane & 15;
    const int g    = lane >> 4;
    const int h    = blockIdx.y;
    const int b    = blockIdx.z;
    const int qrow = blockIdx.x * 128 + wave * 16;
    const size_t rowbase = (size_t)(b * S_ + qrow);
    const int hc = h * DEPTH_;

    // Q row-tile (16 x 64) held in registers: 2 A-fragments
    FragAB qf[2];
#pragma unroll
    for (int f = 0; f < 2; ++f) {
        const unsigned short* p =
            Q + (rowbase + lm) * D_ + hc + f * 32 + g * 8;
        qf[f].q[0] = *(const uint4*)p;
        qf[f].q[1] = *(const uint4*)(p + 16);
    }

    v8f o[4];
#pragma unroll
    for (int dt = 0; dt < 4; ++dt) o[dt] = vzero8();
    float mrow[8], lrow[8];
#pragma unroll
    for (int vr = 0; vr < 8; ++vr) { mrow[vr] = -3.0e38f; lrow[vr] = 0.0f; }

    const int lr = tid >> 2;          // 0..63 key within chunk
    const int lc = (tid & 3) * 16;    // depth column chunk

    for (int j = 0; j < S_; j += KC) {
        __syncthreads();
        {   // stage K chunk (async to LDS) and V chunk (transposed by hand)
            const unsigned short* pk =
                Kmat + (size_t)(b * S_ + j + lr) * D_ + hc + lc;
            unsigned lk = (unsigned)(uintptr_t)&sK[lr * SKW + lc];
            async_ld_b128(lk,      pk);
            async_ld_b128(lk + 16, pk + 8);

            const unsigned short* pv =
                V + (size_t)(b * S_ + j + lr) * D_ + hc + lc;
            uint4 v0 = *(const uint4*)pv;
            uint4 v1 = *(const uint4*)(pv + 8);
            const unsigned short* e0 = (const unsigned short*)&v0;
            const unsigned short* e1 = (const unsigned short*)&v1;
#pragma unroll
            for (int i = 0; i < 8; ++i) sVT[(lc + i)     * SKW + lr] = e0[i];
#pragma unroll
            for (int i = 0; i < 8; ++i) sVT[(lc + 8 + i) * SKW + lr] = e1[i];
        }
        wait_async0();
        __syncthreads();

        // prefetch next chunk's K/V rows into cache while we compute
        if (j + KC < S_) {
            __builtin_prefetch(Kmat + (size_t)(b * S_ + j + KC + lr) * D_ + hc + lc, 0, 1);
            __builtin_prefetch(V    + (size_t)(b * S_ + j + KC + lr) * D_ + hc + lc, 0, 1);
        }

        // ---- logits S = Q * K^T  (16 x 64) ----
        v8f st[4];
#pragma unroll
        for (int nt = 0; nt < 4; ++nt) st[nt] = vzero8();
#pragma unroll
        for (int nt = 0; nt < 4; ++nt) {
#pragma unroll
            for (int f = 0; f < 2; ++f) {
                FragAB kf;
                const int base = (nt * 16 + lm) * SKW + f * 32 + g * 8;
                kf.q[0] = *(const uint4*)&sK[base];
                kf.q[1] = *(const uint4*)&sK[base + 16];
                st[nt] = __builtin_amdgcn_wmma_f32_16x16x32_bf16(
                    false, qf[f].v, false, kf.v,
                    (short)0, st[nt], false, false);
            }
        }
#pragma unroll
        for (int nt = 0; nt < 4; ++nt)
#pragma unroll
            for (int vr = 0; vr < 8; ++vr) st[nt][vr] *= 0.125f;

        // ---- online softmax (rows split per the C-layout lane halves) ----
        float av[8];
#pragma unroll
        for (int vr = 0; vr < 8; ++vr) {
            float r = fmaxf(fmaxf(st[0][vr], st[1][vr]),
                            fmaxf(st[2][vr], st[3][vr]));
            r = fmaxf(r, __shfl_xor(r, 1, 32));
            r = fmaxf(r, __shfl_xor(r, 2, 32));
            r = fmaxf(r, __shfl_xor(r, 4, 32));
            r = fmaxf(r, __shfl_xor(r, 8, 32));
            const float mo = mrow[vr];
            const float mn = fmaxf(mo, r);
            const float al = __expf(mo - mn);
            mrow[vr] = mn;
            float rs = 0.0f;
#pragma unroll
            for (int nt = 0; nt < 4; ++nt) {
                float pp = __expf(st[nt][vr] - mn);
                st[nt][vr] = pp;
                rs += pp;
            }
            rs += __shfl_xor(rs, 1, 32);
            rs += __shfl_xor(rs, 2, 32);
            rs += __shfl_xor(rs, 4, 32);
            rs += __shfl_xor(rs, 8, 32);
            lrow[vr] = lrow[vr] * al + rs;
            av[vr] = al;
        }
#pragma unroll
        for (int dt = 0; dt < 4; ++dt)
#pragma unroll
            for (int vr = 0; vr < 8; ++vr) o[dt][vr] *= av[vr];

        // ---- reshape P: C-layout regs -> per-wave LDS (row-major) ----
#pragma unroll
        for (int nt = 0; nt < 4; ++nt)
#pragma unroll
            for (int vr = 0; vr < 8; ++vr)
                sP[wave][(vr + 8 * g) * SKW + nt * 16 + lm] =
                    f2bf(st[nt][vr]);
        // per-wave LDS ops are in-order: no workgroup barrier needed

        // ---- O += P * V ----
#pragma unroll
        for (int f = 0; f < 2; ++f) {
            FragAB pf;
            const int pb = lm * SKW + f * 32 + g * 8;
            pf.q[0] = *(const uint4*)&sP[wave][pb];
            pf.q[1] = *(const uint4*)&sP[wave][pb + 16];
#pragma unroll
            for (int dt = 0; dt < 4; ++dt) {
                FragAB vf;
                const int vb = (dt * 16 + lm) * SKW + f * 32 + g * 8;
                vf.q[0] = *(const uint4*)&sVT[vb];
                vf.q[1] = *(const uint4*)&sVT[vb + 16];
                o[dt] = __builtin_amdgcn_wmma_f32_16x16x32_bf16(
                    false, pf.v, false, vf.v,
                    (short)0, o[dt], false, false);
            }
        }
    }

    // ---- normalize and emit context (bf16) ----
    float inv[8];
#pragma unroll
    for (int vr = 0; vr < 8; ++vr) inv[vr] = 1.0f / lrow[vr];
#pragma unroll
    for (int dt = 0; dt < 4; ++dt)
#pragma unroll
        for (int vr = 0; vr < 8; ++vr)
            Ctx[(rowbase + vr + 8 * g) * D_ + hc + dt * 16 + lm] =
                f2bf(o[dt][vr] * inv[vr]);
}

// ---------------------------------------------------------------------------
// Host-side launch
// ---------------------------------------------------------------------------
extern "C" void kernel_launch(void* const* d_in, const int* in_sizes, int n_in,
                              void* d_out, int out_size, void* d_ws, size_t ws_size,
                              hipStream_t stream) {
    const float* x  = (const float*)d_in[0];
    const float* wq = (const float*)d_in[1];
    const float* wk = (const float*)d_in[2];
    const float* wv = (const float*)d_in[3];
    const float* wo = (const float*)d_in[4];
    float* out = (float*)d_out;

    const size_t XN = (size_t)M_ * D_;   // 4,194,304 elems
    const size_t WN = (size_t)D_ * D_;   // 1,048,576 elems

    char* ws = (char*)d_ws;
    unsigned short* Xb  = (unsigned short*)ws;  ws += XN * 2;  // reused as Ctx
    unsigned short* WqT = (unsigned short*)ws;  ws += WN * 2;
    unsigned short* WkT = (unsigned short*)ws;  ws += WN * 2;
    unsigned short* WvT = (unsigned short*)ws;  ws += WN * 2;
    unsigned short* WoT = (unsigned short*)ws;  ws += WN * 2;
    unsigned short* Qb  = (unsigned short*)ws;  ws += XN * 2;
    unsigned short* Kb  = (unsigned short*)ws;  ws += XN * 2;
    unsigned short* Vb  = (unsigned short*)ws;  ws += XN * 2;
    unsigned short* Cx  = Xb;  // alias: X no longer needed after QKV GEMMs

    // 1) precision conversion / weight transpose
    cvt_bf16<<<(int)((XN + 255) / 256), 256, 0, stream>>>(x, Xb, (int)XN);
    dim3 tb(32, 8), tg(32, 32);
    cvt_wT<<<tg, tb, 0, stream>>>(wq, WqT);
    cvt_wT<<<tg, tb, 0, stream>>>(wk, WkT);
    cvt_wT<<<tg, tb, 0, stream>>>(wv, WvT);
    cvt_wT<<<tg, tb, 0, stream>>>(wo, WoT);

    // 2) QKV projections
    dim3 gg(D_ / GT_N, M_ / GT_M);   // (8, 32)
    gemm_bf16_wmma<<<gg, 256, 0, stream>>>(Xb, WqT, nullptr, Qb, M_, D_, D_, 0);
    gemm_bf16_wmma<<<gg, 256, 0, stream>>>(Xb, WkT, nullptr, Kb, M_, D_, D_, 0);
    gemm_bf16_wmma<<<gg, 256, 0, stream>>>(Xb, WvT, nullptr, Vb, M_, D_, D_, 0);

    // 3) flash attention
    dim3 ag(S_ / 128, H_, B_);       // (16, 16, 2)
    attn_wmma<<<ag, 256, 0, stream>>>(Qb, Kb, Vb, Cx);

    // 4) output projection -> fp32 result
    gemm_bf16_wmma<<<gg, 256, 0, stream>>>(Cx, WoT, out, nullptr, M_, D_, D_, 1);
}
